// VNETDetector_54211077210247
// MI455X (gfx1250) — matrix operand order repaired and checked by reading, not compiled, not run on hardware
//
#include <hip/hip_runtime.h>

// ---------------------------------------------------------------------------
// VNET detector for MI455X (gfx1250, wave32, WMMA).
//
// Per-sample MLP (1 ->100 sigmoid ->50 relu ->16) over B*L = 2M samples,
// then a 16-state Viterbi scan along L.
// Dominant cost: layer2 GEMM  M=2M, K=100, N=50 (~20 GFLOP) on
// v_wmma_f32_16x16x32_f16, plus 400M v_exp/v_rcp for the sigmoids.
// Priors are produced/consumed chunk-by-chunk (CHUNK=128 -> 8.4 MB) so the
// intermediate stays in the 192 MB L2 instead of a 128 MB HBM round trip.
//
// This revision restructures layer2 kc-outer/nt-inner and caps registers
// via __launch_bounds__(256, 4) so the allocator stays under 256 VGPRs
// (no s_set_vgpr_msb churn, >=4 waves/SIMD to hide trans/WMMA latency).
// ---------------------------------------------------------------------------

#define H1 100
#define H2 50
#define NS 16
#define BB 1024
#define LL 2048
#define CHUNK 128
#define NCHUNK (LL / CHUNK)          // 16
#define ROWS (BB * CHUNK)            // 131072 MLP rows per chunk
#define NTILES (ROWS / 16)           // 8192 16-row tiles per chunk

typedef __attribute__((ext_vector_type(16))) _Float16 v16h;
typedef __attribute__((ext_vector_type(8)))  float    v8f;

__device__ __forceinline__ float fast_sigmoid(float x) {
  // sigmoid(x) = rcp(1 + 2^(-x*log2e)) : one v_exp_f32 + one v_rcp_f32
  float e = __builtin_amdgcn_exp2f(-1.44269504088896f * x);
  return __builtin_amdgcn_rcpf(1.0f + e);
}

__device__ __forceinline__ float lane_gather(float v, int srcLane) {
  int r = __builtin_amdgcn_ds_bpermute(srcLane << 2, __builtin_bit_cast(int, v));
  return __builtin_bit_cast(float, r);
}
__device__ __forceinline__ int lane_gather_i(int v, int srcLane) {
  return __builtin_amdgcn_ds_bpermute(srcLane << 2, v);
}

// 16-bit A fragment (16x32, ISA 7.12.2): lane 0-15 = row M, holds K {0..7,16..23};
// lanes 16-31 hold K {8..15,24..31}.  VGPR slot j packs (k, k+1).
__device__ __forceinline__ int kA(int j, int half) {
  return ((j < 4) ? (2 * j) : (16 + 2 * (j - 4))) + (half ? 8 : 0);
}
// 16-bit B fragment (32x16): lane%16 = col N; same K order as A:
// lanes 0-15 hold K {0..7,16..23}, lanes 16-31 hold K {8..15,24..31}.
__device__ __forceinline__ int kB(int j, int half) {
  return ((j < 4) ? (2 * j) : (16 + 2 * (j - 4))) + (half ? 8 : 0);
}

// ---------------------------------------------------------------------------
// Phase 1: priors for one L-chunk.  One wave handles one 16-row tile pass.
//   layer1 -> one A frag per K-chunk (16 sigmoids/lane), consumed immediately
//   layer2 -> 16x WMMA f16 (K 100->128 padded, N 50->64 padded), bias in C
//   layer3 -> 2x WMMA after per-wave LDS C->A relayout, bias in C
// ---------------------------------------------------------------------------
__global__ __launch_bounds__(256, 4) void vnet_priors_kernel(
    const float* __restrict__ y,  const float* __restrict__ w1,
    const float* __restrict__ b1, const float* __restrict__ w2,
    const float* __restrict__ b2, const float* __restrict__ w3,
    const float* __restrict__ b3, float* __restrict__ pri, int chunk) {
  __shared__ alignas(16) _Float16 w2frag[16][32][16];  // [nt*4+kc][lane][elem] 16KB
  __shared__ alignas(16) _Float16 w3frag[2][32][16];   // 2KB
  __shared__ alignas(16) _Float16 h2s[8][16][64];      // per-wave h2 relayout, 16KB
  __shared__ float w1s[128], b1s[128], b2s[64], b3s[16];

  const int tid  = threadIdx.x;
  const int lane = tid & 31;
  const int wave = tid >> 5;
  const int half = (lane >> 4) & 1;
  const int m    = lane & 15;

  // ---- stage weights into LDS in WMMA fragment layout (once per WG) ----
  for (int rep = 0; rep < 2; ++rep) {          // 512 (frag,lane) slots, 256 thr
    int slot = tid + rep * 256;
    int f = slot >> 5, ln = slot & 31;
    int nt = f >> 2, kc = f & 3;
    int n = nt * 16 + (ln & 15), hh = (ln >> 4) & 1;
    for (int j = 0; j < 8; ++j) {
      int k0 = kc * 32 + kB(j, hh);
      _Float16 v0 = (n < H2 && k0     < H1) ? (_Float16)w2[n * H1 + k0    ] : (_Float16)0.0f;
      _Float16 v1 = (n < H2 && k0 + 1 < H1) ? (_Float16)w2[n * H1 + k0 + 1] : (_Float16)0.0f;
      w2frag[f][ln][2 * j]     = v0;
      w2frag[f][ln][2 * j + 1] = v1;
    }
  }
  if (tid < 64) {                              // 2 w3 frags x 32 lanes
    int f = tid >> 5, ln = tid & 31;
    int n = ln & 15, hh = (ln >> 4) & 1;
    for (int j = 0; j < 8; ++j) {
      int k0 = f * 32 + kB(j, hh);
      w3frag[f][ln][2 * j]     = (k0     < H2) ? (_Float16)w3[n * H2 + k0    ] : (_Float16)0.0f;
      w3frag[f][ln][2 * j + 1] = (k0 + 1 < H2) ? (_Float16)w3[n * H2 + k0 + 1] : (_Float16)0.0f;
    }
  }
  if (tid < 128) { w1s[tid] = (tid < H1) ? w1[tid] : 0.0f;
                   b1s[tid] = (tid < H1) ? b1[tid] : 0.0f; }
  if (tid < 64)  { b2s[tid] = (tid < H2) ? b2[tid] : 0.0f; }
  if (tid < 16)  { b3s[tid] = b3[tid]; }
  __syncthreads();

  const int gwave  = blockIdx.x * 8 + wave;
  const int nwaves = gridDim.x * 8;
  for (int tile = gwave; tile < NTILES; tile += nwaves) {
    const int r0 = tile * 16;
    // chunk row r -> y[b][chunk*CHUNK + t]
    const int r  = r0 + m;
    const float ym = y[(r >> 7) * LL + chunk * CHUNK + (r & 127)];

    // ---- layer 2 accumulators carry the bias (C operand) ----
    v8f acc[4];
#pragma unroll
    for (int nt = 0; nt < 4; ++nt) {
      float bias = b2s[nt * 16 + m];
      acc[nt] = (v8f){bias, bias, bias, bias, bias, bias, bias, bias};
    }

    // ---- fused layer1+layer2: build one A K-chunk, feed 4 N-tiles ----
#pragma unroll
    for (int kc = 0; kc < 4; ++kc) {
      v16h a;
#pragma unroll
      for (int j = 0; j < 8; ++j) {
        int k0 = kc * 32 + kA(j, half);
        float x0 = ym * w1s[k0] + b1s[k0];
        float x1 = ym * w1s[k0 + 1] + b1s[k0 + 1];
        a[2 * j]     = (k0     < H1) ? (_Float16)fast_sigmoid(x0) : (_Float16)0.0f;
        a[2 * j + 1] = (k0 + 1 < H1) ? (_Float16)fast_sigmoid(x1) : (_Float16)0.0f;
      }
#pragma unroll
      for (int nt = 0; nt < 4; ++nt) {
        v16h bf = *(const v16h*)&w2frag[nt * 4 + kc][lane][0];
        acc[nt] = __builtin_amdgcn_wmma_f32_16x16x32_f16(
            false, a, false, bf, (short)0, acc[nt], false, false);
      }
    }

    // ---- relu + C-layout -> A-layout relayout of h2 via per-wave LDS ----
    // C layout: lane holds column n = lane%16, rows v + 8*half.
#pragma unroll
    for (int nt = 0; nt < 4; ++nt) {
      int n = nt * 16 + m;
#pragma unroll
      for (int v = 0; v < 8; ++v) {
        float val = acc[nt][v];
        h2s[wave][v + 8 * half][n] = (_Float16)(val > 0.0f ? val : 0.0f);
      }
    }

    // ---- layer 3: 2 WMMAs (K 50->64 padded), bias in C ----
    // same-wave LDS ops are in order; no cross-wave sharing => no barrier
    float bias3 = b3s[m];
    v8f acc3 = (v8f){bias3, bias3, bias3, bias3, bias3, bias3, bias3, bias3};
#pragma unroll
    for (int c = 0; c < 2; ++c) {
      v16h a3;
#pragma unroll
      for (int j = 0; j < 8; ++j) {
        int k0 = c * 32 + kA(j, half);
        a3[2 * j]     = h2s[wave][m][k0];
        a3[2 * j + 1] = h2s[wave][m][k0 + 1];
      }
      v16h bf = *(const v16h*)&w3frag[c][lane][0];
      acc3 = __builtin_amdgcn_wmma_f32_16x16x32_f16(
          false, a3, false, bf, (short)0, acc3, false, false);
    }

    // ---- store 16x16 priors tile (contiguous 1KB region) ----
#pragma unroll
    for (int v = 0; v < 8; ++v)
      pri[(r0 + v + 8 * half) * NS + m] = acc3[v];
  }
}

// ---------------------------------------------------------------------------
// Phase 2: Viterbi scan over one L-chunk.  lane = state, 2 batch rows/wave.
// Reads the priors chunk (L2-resident), carries in_prob across chunks in ws.
// ---------------------------------------------------------------------------
__global__ __launch_bounds__(256) void vnet_scan_kernel(
    const float* __restrict__ pri, const int* __restrict__ tt,
    float* __restrict__ state, float* __restrict__ out, int chunk) {
  __shared__ float bits_lds[8][2][CHUNK];  // 8KB

  const int tid  = threadIdx.x;
  const int lane = tid & 31;
  const int wave = tid >> 5;
  const int rw   = lane >> 4;     // which of the 2 rows in this wave
  const int s    = lane & 15;     // state
  const int base = rw * 16;       // first lane of this row's half-wave
  const int b    = blockIdx.x * 16 + wave * 2 + rw;

  const int t0 = tt[s * 2 + 0];   // (2s)%16
  const int t1 = tt[s * 2 + 1];   // (2s+1)%16

  float ip = (chunk == 0) ? 0.0f : state[b * NS + s];

  for (int k = 0; k < CHUNK; ++k) {
    float p = pri[(b * CHUNK + k) * NS + s];

    // argmin over 16 states (tie -> lowest index, matching jnp.argmin):
    // butterfly (value,index) reduction with ds_bpermute, stays inside the half
    float mv = ip;
    int   mi = s;
#pragma unroll
    for (int off = 1; off < 16; off <<= 1) {
      int src = lane ^ off;
      float ov = lane_gather(mv, src);
      int   oi = lane_gather_i(mi, src);
      bool take = (ov < mv) || (ov == mv && oi < mi);
      mv = take ? ov : mv;
      mi = take ? oi : mi;
    }
    float bit = (float)(mi & 1);

    float cost = ip - p;
    float c0 = lane_gather(cost, base + t0);
    float c1 = lane_gather(cost, base + t1);
    ip = fminf(c0, c1);

    if (s == 0) bits_lds[wave][rw][k] = bit;
  }
  state[b * NS + s] = ip;

  // coalesced writeback of this wave's 2*CHUNK bits
  const int b0 = blockIdx.x * 16 + wave * 2;
  for (int i = lane; i < 2 * CHUNK; i += 32) {
    int rr = i >> 7, kk = i & 127;
    out[(b0 + rr) * LL + chunk * CHUNK + kk] = bits_lds[wave][rr][kk];
  }
}

// ---------------------------------------------------------------------------
// ws layout: [0, ROWS*NS*4)      priors chunk   (8.39 MB)
//            [.., + BB*NS*4)     carry state    (64 KB)
// ---------------------------------------------------------------------------
extern "C" void kernel_launch(void* const* d_in, const int* in_sizes, int n_in,
                              void* d_out, int out_size, void* d_ws, size_t ws_size,
                              hipStream_t stream) {
  (void)in_sizes; (void)n_in; (void)out_size; (void)ws_size;
  const float* y  = (const float*)d_in[0];
  const float* w1 = (const float*)d_in[1];
  const float* b1 = (const float*)d_in[2];
  const float* w2 = (const float*)d_in[3];
  const float* b2 = (const float*)d_in[4];
  const float* w3 = (const float*)d_in[5];
  const float* b3 = (const float*)d_in[6];
  const int*   tt = (const int*)d_in[7];
  float* out = (float*)d_out;

  float* pri   = (float*)d_ws;
  float* state = (float*)((char*)d_ws + (size_t)ROWS * NS * sizeof(float));

  for (int c = 0; c < NCHUNK; ++c) {
    // 512 WGs x 8 waves = 4096 waves -> 2 tiles/wave (amortizes LDS staging)
    vnet_priors_kernel<<<512, 256, 0, stream>>>(y, w1, b1, w2, b2, w3, b3, pri, c);
    // 64 WGs x 8 waves, 2 rows/wave -> all 1024 batch rows
    vnet_scan_kernel<<<BB / 16, 256, 0, stream>>>(pri, tt, state, out, c);
  }
}
